// LSTM_25451976196279
// MI455X (gfx1250) — matrix operand order repaired and checked by reading, not compiled, not run on hardware
//
#include <hip/hip_runtime.h>
#include <hip/hip_bf16.h>

typedef _Float16 f16;
typedef __attribute__((ext_vector_type(16))) _Float16 v16h;
typedef __attribute__((ext_vector_type(8)))  _Float16 v8h;
typedef __attribute__((ext_vector_type(8)))  float    v8f;

constexpr int Hc = 1024;   // hidden
constexpr int Bc = 32;     // batch
constexpr int Tc = 512;    // time
constexpr int Gc = 4096;   // 4*H

__device__ __forceinline__ float sigmoidf_(float x) { return 1.0f / (1.0f + __expf(-x)); }
__device__ __forceinline__ float tanhf_(float x)    { return 2.0f / (1.0f + __expf(-2.0f * x)) - 1.0f; }

// ---------------------------------------------------------------------------
// Fused LSTM step. One wave per (16 batch rows) x (16 h-units), owning all
// four gate strips for that block:
//
//   gate_g[b][j] = sum_k Ax[b][k]*W_ih[g*H+j][k] + sum_k Hp[b][k]*W_hh[g*H+j][k]
//
// A fragment (16-bit A 16x32): lane l holds row M=l%16,
//   K = kb + (l>>4)*16 .. +16 (contiguous, matching B).
// B fragment (32x16, B(k,n)=W[n][k], W row-major [N][K]): lane l holds
//   col N=l%16 (a W row), K = kb + (l>>4)*16 .. +16 (contiguous 32 bytes).
// C/D layout: VGPR r, lane l -> (M = r + 8*(l>>4), N = l%16), identical for
// all four gate accumulators -> LSTM cell update is elementwise across the
// four accumulators, done in-register. No gates buffer, no second kernel,
// no cross-wave dependency inside a step.
// Grid: 128 blocks x 32 threads = (64 j-strips) x (2 M-tiles).
// ---------------------------------------------------------------------------
__global__ __launch_bounds__(32) void lstm_step(
    const f16* __restrict__ Ax,    // [32][1024] layer input at time t (f16)
    const f16* __restrict__ Hp,    // [32][1024] h_{t-1} (f16)
    const f16* __restrict__ Wih,   // [4096][1024] f16
    const f16* __restrict__ Whh,   // [4096][1024] f16
    const float* __restrict__ bias,// [4096] (b_ih + b_hh)
    float* __restrict__ c,         // [32][1024] cell state (read/write)
    f16* __restrict__ hout,        // [32][1024] h_t (f16) for next step / next layer
    float* __restrict__ out,       // layer1: d_out [B][T][H]; layer0: nullptr
    float* __restrict__ hlast,     // t==T-1: h_last slot [32][1024]; else nullptr
    int t)
{
    const int lane  = threadIdx.x & 31;
    const int lm    = lane & 15;
    const int lg    = lane >> 4;
    const int jbase = (blockIdx.x >> 1) * 16;   // 64 strips of 16 h-units
    const int mt    = blockIdx.x & 1;           // M-tile: batch rows mt*16..mt*16+15

    v8f acc[4] = {{}, {}, {}, {}};              // gates i,f,g,o for this M-tile

    #pragma unroll
    for (int phase = 0; phase < 2; ++phase) {
        const f16* __restrict__ A = phase ? Hp : Ax;
        const f16* __restrict__ W = phase ? Whh : Wih;
        const f16* __restrict__ a0 = A + (size_t)(mt * 16 + lm) * Hc;
        const f16* wp0 = W + (size_t)(0 * Hc + jbase + lm) * Hc;
        const f16* wp1 = W + (size_t)(1 * Hc + jbase + lm) * Hc;
        const f16* wp2 = W + (size_t)(2 * Hc + jbase + lm) * Hc;
        const f16* wp3 = W + (size_t)(3 * Hc + jbase + lm) * Hc;
        #pragma unroll 2
        for (int kb = 0; kb < Hc; kb += 32) {
            const int ka = kb + lg * 16;
            v16h af = *(const v16h*)(a0 + ka);
            const int kw = kb + lg * 16;
            v16h b0 = *(const v16h*)(wp0 + kw);
            v16h b1 = *(const v16h*)(wp1 + kw);
            v16h b2 = *(const v16h*)(wp2 + kw);
            v16h b3 = *(const v16h*)(wp3 + kw);
            acc[0] = __builtin_amdgcn_wmma_f32_16x16x32_f16(false, af, false, b0, (short)0, acc[0], false, false);
            acc[1] = __builtin_amdgcn_wmma_f32_16x16x32_f16(false, af, false, b1, (short)0, acc[1], false, false);
            acc[2] = __builtin_amdgcn_wmma_f32_16x16x32_f16(false, af, false, b2, (short)0, acc[2], false, false);
            acc[3] = __builtin_amdgcn_wmma_f32_16x16x32_f16(false, af, false, b3, (short)0, acc[3], false, false);
        }
    }

    // ---- in-register LSTM cell update (wave-local: all 4 gates present) ----
    const int j = jbase + lm;
    const float b_i = bias[j];
    const float b_f = bias[Hc + j];
    const float b_g = bias[2 * Hc + j];
    const float b_o = bias[3 * Hc + j];

    #pragma unroll
    for (int r = 0; r < 8; ++r) {
        const int m = r + 8 * lg + 16 * mt;
        const size_t idx = (size_t)m * Hc + j;
        const float gi = acc[0][r] + b_i;
        const float gf = acc[1][r] + b_f;
        const float gg = acc[2][r] + b_g;
        const float go = acc[3][r] + b_o;
        const float cn = sigmoidf_(gf) * c[idx] + sigmoidf_(gi) * tanhf_(gg);
        c[idx] = cn;
        const float h = sigmoidf_(go) * tanhf_(cn);
        hout[idx] = (f16)h;
        if (out)   out[(size_t)m * Tc * Hc + (size_t)t * Hc + j] = h;
        if (hlast) hlast[idx] = h;
    }
}

// ---------------------------------------------------------------------------
// Prep / finalize kernels
// ---------------------------------------------------------------------------
__global__ void k_f32_to_f16(const float* __restrict__ in, f16* __restrict__ out) {
    const size_t idx = (size_t)blockIdx.x * 256 + threadIdx.x;
    out[idx] = (f16)in[idx];
}

// x: [B][T][H] f32  ->  xt: [T][B][H] f16
__global__ void k_prep_x(const float* __restrict__ x, f16* __restrict__ xt) {
    const size_t idx = (size_t)blockIdx.x * 256 + threadIdx.x; // over T*B*H
    const int d = idx & (Hc - 1);
    const size_t tb = idx >> 10;
    const int b = (int)(tb & (Bc - 1));
    const int t = (int)(tb >> 5);
    xt[idx] = (f16)x[((size_t)b * Tc + t) * Hc + d];
}

__global__ void k_bias_sum(const float* __restrict__ a, const float* __restrict__ b,
                           float* __restrict__ o) {
    const int idx = blockIdx.x * 256 + threadIdx.x;
    o[idx] = a[idx] + b[idx];
}

// idx over 2*B*H: initialize c (f32) and h_prev (f16) from c0/h0 inputs
__global__ void k_init_state(const float* __restrict__ h0, const float* __restrict__ c0,
                             float* __restrict__ cws, f16* __restrict__ hprev) {
    const int idx = blockIdx.x * 256 + threadIdx.x;
    cws[idx]   = c0[idx];
    hprev[idx] = (f16)h0[idx];
}

// c_last: copy cws [2][B][H] into the tail of d_out (after h_last)
__global__ void k_finalize(const float* __restrict__ cws, float* __restrict__ ctail) {
    const int idx = blockIdx.x * 256 + threadIdx.x;      // over 2*B*H
    ctail[idx] = cws[idx];
}

// ---------------------------------------------------------------------------
extern "C" void kernel_launch(void* const* d_in, const int* in_sizes, int n_in,
                              void* d_out, int out_size, void* d_ws, size_t ws_size,
                              hipStream_t stream)
{
    (void)in_sizes; (void)n_in; (void)out_size; (void)ws_size;

    const float* x     = (const float*)d_in[0];   // [B][T][H]
    const float* h0    = (const float*)d_in[1];   // [2][B][H]
    const float* c0    = (const float*)d_in[2];   // [2][B][H]
    const float* Wih0f = (const float*)d_in[3];   // [4H][H]
    const float* Whh0f = (const float*)d_in[4];
    const float* bih0  = (const float*)d_in[5];
    const float* bhh0  = (const float*)d_in[6];
    const float* Wih1f = (const float*)d_in[7];
    const float* Whh1f = (const float*)d_in[8];
    const float* bih1  = (const float*)d_in[9];
    const float* bhh1  = (const float*)d_in[10];
    float* out = (float*)d_out;

    char* ws = (char*)d_ws;
    const size_t szW  = (size_t)Gc * Hc * sizeof(f16);      // 8 MB
    const size_t szTS = (size_t)Tc * Bc * Hc * sizeof(f16); // 32 MB
    size_t off = 0;
    f16* Wih0 = (f16*)(ws + off); off += szW;
    f16* Whh0 = (f16*)(ws + off); off += szW;
    f16* Wih1 = (f16*)(ws + off); off += szW;
    f16* Whh1 = (f16*)(ws + off); off += szW;
    f16* xh   = (f16*)(ws + off); off += szTS;   // x, time-major f16
    f16* hs0  = (f16*)(ws + off); off += szTS;   // layer0 h sequence f16
    f16* hs1  = (f16*)(ws + off); off += szTS;   // layer1 h sequence f16
    f16* hprev = (f16*)(ws + off); off += (size_t)2 * Bc * Hc * sizeof(f16); // [2][B][H]
    float* bias0 = (float*)(ws + off); off += (size_t)Gc * sizeof(float);
    float* bias1 = (float*)(ws + off); off += (size_t)Gc * sizeof(float);
    float* cws   = (float*)(ws + off); off += (size_t)2 * Bc * Hc * sizeof(float);

    const int nW  = Gc * Hc;          // 4194304
    const int nX  = Tc * Bc * Hc;     // 16777216
    const int nS  = 2 * Bc * Hc;      // 65536
    const size_t stepHB = (size_t)Bc * Hc;
    float* tail = out + (size_t)Bc * Tc * Hc;    // [h_last(2BH)][c_last(2BH)]

    // ---- prep (parallel, one-time per launch) ----
    k_f32_to_f16<<<nW / 256, 256, 0, stream>>>(Wih0f, Wih0);
    k_f32_to_f16<<<nW / 256, 256, 0, stream>>>(Whh0f, Whh0);
    k_f32_to_f16<<<nW / 256, 256, 0, stream>>>(Wih1f, Wih1);
    k_f32_to_f16<<<nW / 256, 256, 0, stream>>>(Whh1f, Whh1);
    k_prep_x<<<nX / 256, 256, 0, stream>>>(x, xh);
    k_bias_sum<<<Gc / 256, 256, 0, stream>>>(bih0, bhh0, bias0);
    k_bias_sum<<<Gc / 256, 256, 0, stream>>>(bih1, bhh1, bias1);
    k_init_state<<<nS / 256, 256, 0, stream>>>(h0, c0, cws, hprev);

    // ---- layer 0 recurrence: one fused kernel per step ----
    for (int t = 0; t < Tc; ++t) {
        const f16* hp = t ? (hs0 + (size_t)(t - 1) * stepHB) : hprev;
        lstm_step<<<(Hc / 16) * 2, 32, 0, stream>>>(
            xh + (size_t)t * stepHB, hp, Wih0, Whh0, bias0,
            cws, hs0 + (size_t)t * stepHB,
            nullptr, (t == Tc - 1) ? tail : nullptr, t);
    }

    // ---- layer 1 recurrence (input = layer0 h sequence) ----
    for (int t = 0; t < Tc; ++t) {
        const f16* hp = t ? (hs1 + (size_t)(t - 1) * stepHB) : (hprev + stepHB);
        lstm_step<<<(Hc / 16) * 2, 32, 0, stream>>>(
            hs0 + (size_t)t * stepHB, hp, Wih1, Whh1, bias1,
            cws + Bc * Hc, hs1 + (size_t)t * stepHB,
            out, (t == Tc - 1) ? (tail + Bc * Hc) : nullptr, t);
    }

    // ---- c_last appended after h_last ----
    k_finalize<<<nS / 256, 256, 0, stream>>>(cws, tail + 2 * Bc * Hc);
}
